// RelationalGraphConvolutionRP_65747359367363
// MI455X (gfx1250) — matrix operand order repaired and checked by reading, not compiled, not run on hardware
//
#include <hip/hip_runtime.h>
#include <hip/hip_bf16.h>

#define R0C 8
#define RC  17
#define INC 256
#define OUTC 256
#define NBC 8

typedef __attribute__((ext_vector_type(16))) __bf16 v16bf;
typedef __attribute__((ext_vector_type(8)))  float  v8f;

union ABFrag { v16bf v; uint4 u[2]; };

// ---------------- small helper kernels ----------------

__global__ void zero_i32(int* __restrict__ p, int n) {
  int i = blockIdx.x * blockDim.x + threadIdx.x;
  if (i < n) p[i] = 0;
}

__global__ void init_out(float* __restrict__ out, const float* __restrict__ bias, int n) {
  int i = blockIdx.x * blockDim.x + threadIdx.x;
  if (i < n) out[i] = bias[i & (OUTC - 1)];
}

__global__ void count_edges(const int* __restrict__ t, int* __restrict__ cnt, int E, int N) {
  int e = blockIdx.x * blockDim.x + threadIdx.x;
  if (e >= E) return;
  int s = t[3 * e + 0];
  int r = t[3 * e + 1];
  int o = t[3 * e + 2];
  atomicAdd(&cnt[r * N + o], 1);            // col of forward message
  atomicAdd(&cnt[(r + R0C) * N + s], 1);    // col of reverse message
}

__global__ void f32_to_bf16(const float* __restrict__ x, __bf16* __restrict__ y, int n) {
  int i = blockIdx.x * blockDim.x + threadIdx.x;
  if (i < n) y[i] = (__bf16)x[i];
}

// weights[r][i][o] = sum_b comps[r][b] * bases[b][i][o], emitted as bf16
__global__ void make_weights(const float* __restrict__ bases, const float* __restrict__ comps,
                             __bf16* __restrict__ w) {
  int idx = blockIdx.x * blockDim.x + threadIdx.x;
  const int total = RC * INC * OUTC;
  if (idx >= total) return;
  int o = idx & (OUTC - 1);
  int i = (idx >> 8) & (INC - 1);
  int r = idx >> 16;
  float acc = 0.f;
#pragma unroll
  for (int b = 0; b < NBC; ++b)
    acc += comps[r * NBC + b] * bases[(b * INC + i) * OUTC + o];
  w[idx] = (__bf16)acc;
}

// ---------------- WMMA GEMM: C(MxOUT) = A(MxK,bf16) * B(KxOUT,bf16), f32 accum ----
// One wave computes a 16x32 tile of C (two 16x16 WMMA accumulators).

__global__ void __launch_bounds__(256)
gemm_wmma_bf16(const __bf16* __restrict__ A, const __bf16* __restrict__ Bm,
               float* __restrict__ C, int M) {
  const int K = INC;
  int wave = (int)((blockIdx.x * blockDim.x + threadIdx.x) >> 5);
  int lane = (int)(threadIdx.x & 31);
  const int ntn = OUTC / 32;            // 8 column tiles of width 32
  int mt = wave / ntn;
  int nt = wave - mt * ntn;
  int m0 = mt * 16;
  if (m0 + 16 > M) return;
  int n0 = nt * 32;

  // A fragment addressing (16-bit A 16x32 layout, ISA 7.12.2):
  // lanes 0-15: row=lane, K {0..7,16..23}; lanes 16-31: row=lane-16, K {8..15,24..31}
  int arow  = m0 + (lane & 15);
  int akoff = (lane >> 4) << 3;         // 0 or 8
  const __bf16* abase = A + arow * K + akoff;

  v8f acc0 = {}, acc1 = {};
#pragma unroll 2
  for (int k0 = 0; k0 < K; k0 += 32) {
    ABFrag a, b0, b1;
    a.u[0] = *(const uint4*)(abase + k0);
    a.u[1] = *(const uint4*)(abase + k0 + 16);
    // B fragment: lane holds column N = n0+(lane&15) (b1: +16), K in the same order as A
#pragma unroll
    for (int s = 0; s < 16; ++s) {
      int kk = k0 + akoff + (s & 7) + ((s >> 3) << 4);
      const __bf16* bp = Bm + kk * OUTC + n0 + (lane & 15);
      b0.v[s] = bp[0];
      b1.v[s] = bp[16];
    }
    acc0 = __builtin_amdgcn_wmma_f32_16x16x32_bf16(false, a.v, false, b0.v,
                                                   (short)0, acc0, false, false);
    acc1 = __builtin_amdgcn_wmma_f32_16x16x32_bf16(false, a.v, false, b1.v,
                                                   (short)0, acc1, false, false);
  }

  // C layout: VGPR i -> row m0+i (lanes 0-15) / m0+8+i (lanes 16-31), col n0+(lane&15)
  int crow = m0 + ((lane >> 4) << 3);
  int ccol = n0 + (lane & 15);
#pragma unroll
  for (int i = 0; i < 8; ++i) {
    C[(crow + i) * OUTC + ccol]      = acc0[i];
    C[(crow + i) * OUTC + ccol + 16] = acc1[i];
  }
}

// ---------------- per-relation scatter (wave per edge, 8 channels per lane) -------

__global__ void __launch_bounds__(256)
scatter_rel(const int* __restrict__ t, const int* __restrict__ cnt,
            const float* __restrict__ fwF, const float* __restrict__ fwB,
            float* __restrict__ out, int rp, int E, int N) {
  int lane = (int)(threadIdx.x & 31);
  int gw = (int)((blockIdx.x * blockDim.x + threadIdx.x) >> 5);
  int nw = (int)((gridDim.x * blockDim.x) >> 5);
  int c0 = lane << 3;
  for (int e = gw; e < E; e += nw) {
    if (t[3 * e + 1] != rp) continue;
    int s = t[3 * e + 0];
    int o = t[3 * e + 2];
    float vf = 1.0f / (float)cnt[(rp + R0C) * N + s];  // forward msg norm
    float vb = 1.0f / (float)cnt[rp * N + o];          // reverse msg norm
    const float4* pf = (const float4*)(fwF + o * OUTC + c0);
    const float4* pb = (const float4*)(fwB + s * OUTC + c0);
    float4 f0 = pf[0], f1 = pf[1];
    float4 g0 = pb[0], g1 = pb[1];
    float* os = out + s * OUTC + c0;
    float* oo = out + o * OUTC + c0;
    atomicAdd(os + 0, vf * f0.x); atomicAdd(os + 1, vf * f0.y);
    atomicAdd(os + 2, vf * f0.z); atomicAdd(os + 3, vf * f0.w);
    atomicAdd(os + 4, vf * f1.x); atomicAdd(os + 5, vf * f1.y);
    atomicAdd(os + 6, vf * f1.z); atomicAdd(os + 7, vf * f1.w);
    atomicAdd(oo + 0, vb * g0.x); atomicAdd(oo + 1, vb * g0.y);
    atomicAdd(oo + 2, vb * g0.z); atomicAdd(oo + 3, vb * g0.w);
    atomicAdd(oo + 4, vb * g1.x); atomicAdd(oo + 5, vb * g1.y);
    atomicAdd(oo + 6, vb * g1.z); atomicAdd(oo + 7, vb * g1.w);
  }
}

// self-loop relation: count is always 1, so out += fw_self (stream-ordered, no atomics)
__global__ void add_selfloop(const float* __restrict__ fwS, float* __restrict__ out, int n) {
  int i = blockIdx.x * blockDim.x + threadIdx.x;
  if (i < n) out[i] += fwS[i];
}

// ---------------- launch ----------------

extern "C" void kernel_launch(void* const* d_in, const int* in_sizes, int n_in,
                              void* d_out, int out_size, void* d_ws, size_t ws_size,
                              hipStream_t stream) {
  const int*   triples  = (const int*)d_in[0];
  const float* features = (const float*)d_in[1];
  const float* bases    = (const float*)d_in[2];
  const float* comps    = (const float*)d_in[3];
  const float* bias     = (const float*)d_in[4];
  float* out = (float*)d_out;

  const int E = in_sizes[0] / 3;
  const int N = in_sizes[1] / INC;
  (void)n_in; (void)out_size; (void)ws_size;

  // workspace layout (256B aligned): cnt | feat_bf16 | weights_bf16 | fw_fwd | fw_bwd
  char* ws = (char*)d_ws;
  size_t off = 0;
  auto al = [](size_t x) { return (x + 255) & ~(size_t)255; };
  int*    cnt   = (int*)(ws + off);    off = al(off + (size_t)RC * N * sizeof(int));
  __bf16* featb = (__bf16*)(ws + off); off = al(off + (size_t)N * INC * sizeof(__bf16));
  __bf16* wb    = (__bf16*)(ws + off); off = al(off + (size_t)RC * INC * OUTC * sizeof(__bf16));
  float*  fwF   = (float*)(ws + off);  off = al(off + (size_t)N * OUTC * sizeof(float));
  float*  fwB   = (float*)(ws + off);  off = al(off + (size_t)N * OUTC * sizeof(float));

  const int rn = RC * N;
  zero_i32<<<(rn + 255) / 256, 256, 0, stream>>>(cnt, rn);
  init_out<<<(N * OUTC + 255) / 256, 256, 0, stream>>>(out, bias, N * OUTC);
  count_edges<<<(E + 255) / 256, 256, 0, stream>>>(triples, cnt, E, N);
  f32_to_bf16<<<(N * INC + 255) / 256, 256, 0, stream>>>(features, featb, N * INC);
  make_weights<<<(RC * INC * OUTC + 255) / 256, 256, 0, stream>>>(bases, comps, wb);

  const int mtiles = N / 16;                     // 1875 (N divisible by 16)
  const int gemm_waves = mtiles * (OUTC / 32);   // 15000
  const int gemm_blocks = (gemm_waves * 32 + 255) / 256;

  for (int rp = 0; rp < R0C; ++rp) {
    gemm_wmma_bf16<<<gemm_blocks, 256, 0, stream>>>(featb, wb + (size_t)rp * INC * OUTC, fwF, N);
    gemm_wmma_bf16<<<gemm_blocks, 256, 0, stream>>>(featb, wb + (size_t)(rp + R0C) * INC * OUTC, fwB, N);
    scatter_rel<<<2048, 256, 0, stream>>>(triples, cnt, fwF, fwB, out, rp, E, N);
  }
  // self-loop: W[2*R0], normalization == 1
  gemm_wmma_bf16<<<gemm_blocks, 256, 0, stream>>>(featb, wb + (size_t)(2 * R0C) * INC * OUTC, fwF, N);
  add_selfloop<<<(N * OUTC + 255) / 256, 256, 0, stream>>>(fwF, out, N * OUTC);
}
